// Encoder_16449724744797
// MI455X (gfx1250) — matrix operand order repaired and checked by reading, not compiled, not run on hardware
//
#include <hip/hip_runtime.h>

// Bidirectional GRU encoder for MI455X (gfx1250, wave32, WMMA bf16 path).
//
// VOCAB=32000 EMB=300 UNITS=512 B=64 T=512
//   out = (B,T,2*512) f32  then  state = (B,1024) f32, concatenated flat.
//
// Strategy (latency-bound recurrence):
//   1. Convert emb/W/U to bf16; W,U stored K-contiguous ("transposed") so WMMA
//      A and B fragments are contiguous 16B loads per lane. W/emb K padded
//      300 -> 320.
//   2. xw = x@W + b0 for all (t,b) in one parallel WMMA GEMM (both dirs).
//   3. 512 step kernels (graph replay amortizes launches), each:
//      rec = h@U (WMMA, K=512) into LDS, 6 independent accumulator chains
//      sharing one A fragment, software-pipelined fragment loads (next k-step
//      issued before current WMMAs), global_prefetch of phase-2 xw/h lines
//      overlapping the GEMM; barrier; fused sigmoid/tanh update; double-
//      buffered h (f32 + bf16).
//
// Workspace: ~429 MB (embB 20.5M, WT 2x1M, UT 2x1.5M, h 0.75M, xw 2x201M).

#define BB    64
#define TT    512
#define EMBD  300
#define KP    320      // padded K for the input projection
#define NU    512      // UNITS (recurrent K)
#define N3    1536     // 3*UNITS
#define NVOC  32000

typedef __bf16 bf16;
typedef __attribute__((ext_vector_type(16))) __bf16 v16bf;
typedef __attribute__((ext_vector_type(8)))  __bf16 v8bf;
typedef __attribute__((ext_vector_type(8)))  float  v8f;

static __device__ __forceinline__ v16bf cat8(v8bf lo, v8bf hi) {
  return __builtin_shufflevector(lo, hi, 0,1,2,3,4,5,6,7,8,9,10,11,12,13,14,15);
}

// A fragment (16x32 bf16), ISA 7.12.2: lane = (M&15) + 16*kg.
// halves 0..7  -> K = k0 + 8*kg + {0..7}
// halves 8..15 -> K = k0 + 16 + 8*kg + {0..7}
static __device__ __forceinline__ v16bf load_a_frag(const bf16* row, int k0, int kg) {
  v8bf lo = *(const v8bf*)(row + k0 + 8 * kg);
  v8bf hi = *(const v8bf*)(row + k0 + 16 + 8 * kg);
  return cat8(lo, hi);
}

// B fragment (32x16 bf16): lane = N + 16*kh; halves 0..7 -> K = k0 + 8*kh + h,
// halves 8..15 -> K = k0 + 16 + 8*kh + (h-8)
// (row pointer is into the K-contiguous transposed weight array)
static __device__ __forceinline__ v16bf load_b_frag(const bf16* row, int k0, int kh) {
  v8bf lo = *(const v8bf*)(row + k0 + 8 * kh);
  v8bf hi = *(const v8bf*)(row + k0 + 16 + 8 * kh);
  return cat8(lo, hi);
}

static __device__ __forceinline__ v8f wmma_bf16(v16bf a, v16bf b, v8f c) {
  return __builtin_amdgcn_wmma_f32_16x16x32_bf16(false, a, false, b, (short)0, c,
                                                 false, false);
}

// ---------------------------------------------------------------- converts
__global__ void k_cvt_emb(const float* __restrict__ src, bf16* __restrict__ dst) {
  long i = (long)blockIdx.x * 256 + threadIdx.x;
  if (i >= (long)NVOC * KP) return;
  long r = i / KP;
  int  c = (int)(i - r * KP);
  dst[i] = (c < EMBD) ? (bf16)src[r * EMBD + c] : (bf16)0.0f;
}

__global__ void k_cvt_w(const float* __restrict__ Wf, const float* __restrict__ Wb,
                        bf16* __restrict__ WTf, bf16* __restrict__ WTb) {
  long i = (long)blockIdx.x * 256 + threadIdx.x;
  if (i >= 2L * N3 * KP) return;
  int  d = (int)(i / ((long)N3 * KP));
  long r = i - (long)d * N3 * KP;
  int  n = (int)(r / KP);
  int  k = (int)(r - (long)n * KP);
  const float* W  = d ? Wb : Wf;
  bf16*        WT = d ? WTb : WTf;
  WT[r] = (k < EMBD) ? (bf16)W[(long)k * N3 + n] : (bf16)0.0f;
}

__global__ void k_cvt_u(const float* __restrict__ Uf, const float* __restrict__ Ub,
                        bf16* __restrict__ UTf, bf16* __restrict__ UTb) {
  long i = (long)blockIdx.x * 256 + threadIdx.x;
  if (i >= 2L * N3 * NU) return;
  int  d = (int)(i / ((long)N3 * NU));
  long r = i - (long)d * N3 * NU;
  int  n = (int)(r / NU);
  int  k = (int)(r - (long)n * NU);
  const float* U  = d ? Ub : Uf;
  bf16*        UT = d ? UTb : UTf;
  UT[r] = (bf16)U[(long)k * N3 + n];
}

__global__ void k_init_h(float* __restrict__ hf, bf16* __restrict__ hb) {
  int i = blockIdx.x * 256 + threadIdx.x;
  if (i < 2 * 2 * BB * NU) { hf[i] = 0.0f; hb[i] = (bf16)0.0f; }
}

// ------------------------------------------------- xw = gather(emb) @ W + b0
// Grid: 12288 WGs x 256 thr (8 waves). Wave = one 16-row x 64-col strip,
// K = 320 (10 wmma steps), 4 accumulators sharing one A fragment,
// software-pipelined fragment loads.
__global__ void k_xw_gemm(const int* __restrict__ tokens, const bf16* __restrict__ embB,
                          const bf16* __restrict__ WTf, const bf16* __restrict__ WTb,
                          const float* __restrict__ b_f, const float* __restrict__ b_b,
                          float* __restrict__ xwf, float* __restrict__ xwb) {
  int wave = threadIdx.x >> 5, lane = threadIdx.x & 31;
  int gw = blockIdx.x * 8 + wave;        // 0..98303
  int d  = gw / 49152;                   // direction
  int r  = gw - d * 49152;
  int mT = r / 24;                       // 0..2047 : 16-row tile over m = t*64+b
  int n0 = (r - mT * 24) * 64;           // 64-col strip

  const bf16*  WT   = d ? WTb : WTf;
  const float* bias = d ? b_b : b_f;     // row 0 of b = input bias
  float*       xw   = d ? xwb : xwf;

  int ln = lane & 15, kg = lane >> 4;
  int m = mT * 16 + ln;
  int t = m >> 6, b = m & 63;
  int tok = tokens[b * TT + t];          // tokens is (B,T)
  const bf16* arow = embB + (long)tok * KP;

  const bf16* brow[4];
#pragma unroll
  for (int i = 0; i < 4; ++i) brow[i] = WT + (long)(n0 + i * 16 + ln) * KP;

  v8f acc[4];
#pragma unroll
  for (int i = 0; i < 4; ++i) acc[i] = (v8f){};

  // software pipeline: fragments for k-step k0 live while k0+32 loads issue
  v16bf aCur = load_a_frag(arow, 0, kg);
  v16bf bCur[4];
#pragma unroll
  for (int i = 0; i < 4; ++i) bCur[i] = load_b_frag(brow[i], 0, kg);

  for (int k0 = 0; k0 < KP - 32; k0 += 32) {
    v16bf aNxt = load_a_frag(arow, k0 + 32, kg);
    v16bf bNxt[4];
#pragma unroll
    for (int i = 0; i < 4; ++i) bNxt[i] = load_b_frag(brow[i], k0 + 32, kg);
#pragma unroll
    for (int i = 0; i < 4; ++i) acc[i] = wmma_bf16(aCur, bCur[i], acc[i]);
    aCur = aNxt;
#pragma unroll
    for (int i = 0; i < 4; ++i) bCur[i] = bNxt[i];
  }
#pragma unroll
  for (int i = 0; i < 4; ++i) acc[i] = wmma_bf16(aCur, bCur[i], acc[i]);

  // C/D layout: vgpr v -> row (v + 8*kg), col = lane&15 within tile
#pragma unroll
  for (int v = 0; v < 8; ++v) {
    long ms = (long)mT * 16 + v + 8 * kg;   // = t*64+b of this output row
    float* orow = xw + ms * N3;
    int c = n0 + ln;
    orow[c +  0] = acc[0][v] + bias[c +  0];
    orow[c + 16] = acc[1][v] + bias[c + 16];
    orow[c + 32] = acc[2][v] + bias[c + 32];
    orow[c + 48] = acc[3][v] + bias[c + 48];
  }
}

// ------------------------------------------------------ one GRU time step
// Grid: 16 WGs (2 dirs x 8 column blocks of 64) x 256 thr (8 waves).
// Phase 0: global_prefetch the phase-2 xw slice (48 KB) + h slice (16 KB)
//          so the HBM fetch overlaps the phase-1 GEMM.
// Phase 1: rec = h @ U. Wave -> fixed 16-row A tile (mt = wave>>1) and 6
//          (gate,ntile) outputs sharing the A fragment: 6 independent WMMA
//          chains, K=512, software-pipelined fragment loads, results in LDS.
// Phase 2: fused gate update; writes ys, double-buffered h (f32 + bf16).
__global__ void k_gru_step(const bf16* __restrict__ UTf, const bf16* __restrict__ UTb,
                           const float* __restrict__ xwf, const float* __restrict__ xwb,
                           const float* __restrict__ b_f, const float* __restrict__ b_b,
                           float* __restrict__ hf32, bf16* __restrict__ hbf16,
                           float* __restrict__ out, int s) {
  __shared__ float rec[3][BB][64];       // 48 KB

  int d  = blockIdx.x >> 3;
  int c0 = (blockIdx.x & 7) * 64;
  int buf = s & 1, nbuf = buf ^ 1;

  const bf16*  UT   = d ? UTb : UTf;
  const float* xw   = d ? xwb : xwf;
  const float* bias = d ? b_b : b_f;     // row 1 of b = recurrent bias
  int t_out = d ? (TT - 1 - s) : s;

  const bf16*  hA   = hbf16 + (long)(buf * 2 + d) * BB * NU;
  const float* hOld = hf32  + (long)(buf * 2 + d) * BB * NU;

  // ---- Phase 0: prefetch phase-2 operands (overlaps with GEMM below).
  // xw slice: 64 rows x 3 gate-segments of 64 floats (2 x 128B lines each)
  // -> 384 lines; h slice: 64 rows x 64 floats (2 lines) -> 128 lines.
  {
    int p = threadIdx.x;                                  // 0..255
    int q = p;                                            // line id 0..383
    int b = q / 6, seg = q - b * 6;
    const float* a = xw + ((long)t_out * BB + b) * N3
                        + (seg >> 1) * NU + c0 + (seg & 1) * 32;
    __builtin_prefetch(a, 0, 1);
    q = p + 256;
    if (q < 384) {
      b = q / 6; seg = q - b * 6;
      a = xw + ((long)t_out * BB + b) * N3 + (seg >> 1) * NU + c0 + (seg & 1) * 32;
      __builtin_prefetch(a, 0, 1);
    }
    if (p < 128) {                                        // hOld lines
      int br = p >> 1;
      __builtin_prefetch(hOld + br * NU + c0 + (p & 1) * 32, 0, 1);
    }
  }

  int wave = threadIdx.x >> 5, lane = threadIdx.x & 31;
  int ln = lane & 15, kg = lane >> 4;

  // ---- Phase 1: rec = h @ U
  int mt   = wave >> 1;                  // fixed A row-tile per wave (0..3)
  int half = wave & 1;                   // which 6 of the 12 (g,nt) combos
  const bf16* arow = hA + (long)(mt * 16 + ln) * NU;

  const bf16* brow[6];
#pragma unroll
  for (int i = 0; i < 6; ++i) {
    int idx = half * 6 + i;              // 0..11
    int g = idx >> 2, nt = idx & 3;
    brow[i] = UT + (long)(g * NU + c0 + nt * 16 + ln) * NU;
  }

  v8f acc[6];
#pragma unroll
  for (int i = 0; i < 6; ++i) acc[i] = (v8f){};

  // software pipeline: issue next k-step's 1 A + 6 B loads before the
  // current k-step's 6 WMMAs so loads get a full WMMA group of cover.
  v16bf aCur = load_a_frag(arow, 0, kg);
  v16bf bCur[6];
#pragma unroll
  for (int i = 0; i < 6; ++i) bCur[i] = load_b_frag(brow[i], 0, kg);

  for (int k0 = 0; k0 < NU - 32; k0 += 32) {
    v16bf aNxt = load_a_frag(arow, k0 + 32, kg);
    v16bf bNxt[6];
#pragma unroll
    for (int i = 0; i < 6; ++i) bNxt[i] = load_b_frag(brow[i], k0 + 32, kg);
#pragma unroll
    for (int i = 0; i < 6; ++i) acc[i] = wmma_bf16(aCur, bCur[i], acc[i]);
    aCur = aNxt;
#pragma unroll
    for (int i = 0; i < 6; ++i) bCur[i] = bNxt[i];
  }
#pragma unroll
  for (int i = 0; i < 6; ++i) acc[i] = wmma_bf16(aCur, bCur[i], acc[i]);

#pragma unroll
  for (int i = 0; i < 6; ++i) {
    int idx = half * 6 + i;
    int g = idx >> 2, nt = idx & 3;
    int jj = nt * 16 + ln;
#pragma unroll
    for (int v = 0; v < 8; ++v)
      rec[g][mt * 16 + v + 8 * kg][jj] = acc[i][v];
  }

  __syncthreads();

  // ---- Phase 2: fused gate update
  float* hNewF = hf32  + (long)(nbuf * 2 + d) * BB * NU;
  bf16*  hNewB = hbf16 + (long)(nbuf * 2 + d) * BB * NU;

  for (int i = 0; i < 16; ++i) {
    int idx = threadIdx.x + i * 256;     // 4096 elements: 64 rows x 64 cols
    int b = idx >> 6, j = idx & 63;
    int col = c0 + j;
    const float* xrow = xw + ((long)t_out * BB + b) * N3;
    float xz = xrow[col], xr = xrow[NU + col], xh = xrow[2 * NU + col];
    float rz = rec[0][b][j] + bias[N3 + col];
    float rr = rec[1][b][j] + bias[N3 + NU + col];
    float rh = rec[2][b][j] + bias[N3 + 2 * NU + col];
    float z  = 1.0f / (1.0f + __expf(-(xz + rz)));
    float rg = 1.0f / (1.0f + __expf(-(xr + rr)));
    float hh = tanhf(xh + rg * rh);
    float hp = hOld[b * NU + col];
    float hn = z * hp + (1.0f - z) * hh;
    hNewF[b * NU + col] = hn;
    hNewB[b * NU + col] = (bf16)hn;
    out[((long)b * TT + t_out) * 1024 + d * NU + col] = hn;     // ys
    if (s == TT - 1)
      out[(long)BB * TT * 1024 + (long)b * 1024 + d * NU + col] = hn;  // state
  }
}

// ----------------------------------------------------------------- launch
extern "C" void kernel_launch(void* const* d_in, const int* in_sizes, int n_in,
                              void* d_out, int out_size, void* d_ws, size_t ws_size,
                              hipStream_t stream) {
  const int*   tokens = (const int*)  d_in[0];
  const float* emb    = (const float*)d_in[1];
  const float* W_f    = (const float*)d_in[2];
  const float* U_f    = (const float*)d_in[3];
  const float* b_f    = (const float*)d_in[4];
  const float* W_b    = (const float*)d_in[5];
  const float* U_b    = (const float*)d_in[6];
  const float* b_b    = (const float*)d_in[7];
  float* out = (float*)d_out;

  char* ws = (char*)d_ws;
  size_t off = 0;
  auto carve = [&](size_t bytes) {
    char* p = ws + off;
    off += (bytes + 255) & ~(size_t)255;
    return p;
  };
  bf16*  embB  = (bf16*) carve((size_t)NVOC * KP * sizeof(bf16));
  bf16*  WTf   = (bf16*) carve((size_t)N3 * KP * sizeof(bf16));
  bf16*  WTb   = (bf16*) carve((size_t)N3 * KP * sizeof(bf16));
  bf16*  UTf   = (bf16*) carve((size_t)N3 * NU * sizeof(bf16));
  bf16*  UTb   = (bf16*) carve((size_t)N3 * NU * sizeof(bf16));
  bf16*  hb16  = (bf16*) carve((size_t)2 * 2 * BB * NU * sizeof(bf16));
  float* hf32  = (float*)carve((size_t)2 * 2 * BB * NU * sizeof(float));
  float* xwf   = (float*)carve((size_t)TT * BB * N3 * sizeof(float));
  float* xwb   = (float*)carve((size_t)TT * BB * N3 * sizeof(float));
  (void)ws_size; (void)in_sizes; (void)n_in; (void)out_size;

  k_cvt_emb<<<(NVOC * KP + 255) / 256, 256, 0, stream>>>(emb, embB);
  k_cvt_w  <<<(2 * N3 * KP + 255) / 256, 256, 0, stream>>>(W_f, W_b, WTf, WTb);
  k_cvt_u  <<<(2 * N3 * NU + 255) / 256, 256, 0, stream>>>(U_f, U_b, UTf, UTb);
  k_init_h <<<(2 * 2 * BB * NU + 255) / 256, 256, 0, stream>>>(hf32, hb16);

  k_xw_gemm<<<12288, 256, 0, stream>>>(tokens, embB, WTf, WTb, b_f, b_b, xwf, xwb);

  for (int s = 0; s < TT; ++s)
    k_gru_step<<<16, 256, 0, stream>>>(UTf, UTb, xwf, xwb, b_f, b_b,
                                       hf32, hb16, out, s);
}
